// TransformerEncoder_48928267436428
// MI455X (gfx1250) — matrix operand order repaired and checked by reading, not compiled, not run on hardware
//
#include <hip/hip_runtime.h>
#include <hip/hip_bf16.h>

// ---------------------------------------------------------------------------
// MI455X (gfx1250) fused multi-head attention, bf16 WMMA path.
//   wmma shape: 16x16x32 bf16 -> f32, wave32.
//   A layout: lane L<16 -> row L, elems K = {0..15};
//             lane L>=16 -> row L-16, elems K = {16..31}.
//   B layout: lane L -> col L%16, elems K = {0..15} + 16*(L/16).
//   C/D layout: lane L -> col L%16, vgpr r -> row r + 8*(L/16).
// ---------------------------------------------------------------------------

typedef __bf16 bf16_t;
typedef __attribute__((ext_vector_type(16))) __bf16 v16bf;
typedef __attribute__((ext_vector_type(8)))  __bf16 v8bf;
typedef __attribute__((ext_vector_type(8)))  float  v8f;

#define HEADS 12
#define HD    64
#define NSEQ  197
#define NPAD  208          // 13 * 16
#define FEATS 768
#define BATCH 128
#define QTILES 13          // ceil(197/16)

__device__ __forceinline__ v8f wmma_bf16(v16bf a, v16bf b, v8f c) {
  return __builtin_amdgcn_wmma_f32_16x16x32_bf16(
      /*neg_a=*/false, a, /*neg_b=*/false, b,
      /*c_mod=*/(short)0, c, /*reuse_a=*/false, /*reuse_b=*/false);
}

// ---------------------------------------------------------------------------
// Kernel 0a: W_out [768,768] f32 row-major -> WT [n][k] bf16 (B-operand layout)
// ---------------------------------------------------------------------------
__global__ void wt_kernel(const float* __restrict__ W, bf16_t* __restrict__ WT) {
  int idx = blockIdx.x * blockDim.x + threadIdx.x;     // 768*768 threads
  int k = idx / FEATS;
  int n = idx % FEATS;
  WT[(size_t)n * FEATS + k] = (__bf16)W[idx];
}

// ---------------------------------------------------------------------------
// Kernel 0b: W_qkv [12,64,192] f32 -> WqT [12,192,64] bf16 (B-operand layout)
// ---------------------------------------------------------------------------
__global__ void wqt_kernel(const float* __restrict__ W, bf16_t* __restrict__ WT) {
  int idx = blockIdx.x * blockDim.x + threadIdx.x;     // 12*64*192 threads
  int h   = idx / (HD * 3 * HD);
  int rem = idx % (HD * 3 * HD);
  int k   = rem / (3 * HD);
  int c   = rem % (3 * HD);
  WT[((size_t)h * (3 * HD) + c) * HD + k] = (__bf16)W[idx];
}

// ---------------------------------------------------------------------------
// Kernel 1: per-(b,h) QKV projection, X_h[197,64] x W_qkv[h][64,192].
//   Q,K stored bf16 [bh][NPAD][64] row-major (pad rows = bias).
//   V stored transposed bf16 [bh][64][NPAD] (pad cols = bias).
// ---------------------------------------------------------------------------
__global__ __launch_bounds__(128)
void qkv_kernel(const float* __restrict__ x, const bf16_t* __restrict__ WqT,
                const float* __restrict__ bq,
                bf16_t* __restrict__ Qb, bf16_t* __restrict__ Kb,
                bf16_t* __restrict__ VTb) {
  int bh   = blockIdx.x;                 // b*12 + h
  int b    = bh / HEADS;
  int h    = bh % HEADS;
  int lane = threadIdx.x & 31;
  int wave = threadIdx.x >> 5;
  int ln   = lane & 15;
  int hi   = lane >> 4;

  for (int tile = wave; tile < QTILES * 12; tile += 4) {
    int mt = tile % QTILES;              // 16-row tile of tokens
    int nt = tile / QTILES;              // 16-col tile of the 192 outputs
    int m0 = mt * 16, n0 = nt * 16;
    int c  = n0 + ln;

    v8f acc = {};
    #pragma unroll
    for (int s = 0; s < 2; ++s) {        // contraction over 64 = 2 x 32
      // A operand: x head slice, rows = tokens, fp32 -> bf16 on the fly
      v16bf a = {};
      int tok = m0 + ln;
      if (tok < NSEQ) {
        const float* xp = x + ((size_t)b * NSEQ + tok) * FEATS + h * HD
                            + s * 32 + 16 * hi;
        #pragma unroll
        for (int e = 0; e < 8; ++e) {
          a[e]     = (__bf16)xp[e];          // K = s*32 + 16*hi + e
          a[e + 8] = (__bf16)xp[e + 8];      // K = s*32 + 16*hi + 8 + e
        }
      }
      // B operand: pre-transposed bf16 weights, contiguous 32B per lane
      const bf16_t* wp = WqT + ((size_t)h * (3 * HD) + c) * HD
                             + s * 32 + 16 * hi;
      v16bf bv = *(const v16bf*)wp;
      acc = wmma_bf16(a, bv, acc);
    }

    float bias = bq[h * (3 * HD) + c];

    if (n0 < 64) {                       // ---- Q ----
      int d = c;
      bf16_t* qp = Qb + (size_t)bh * NPAD * HD + d;
      #pragma unroll
      for (int r = 0; r < 8; ++r) {
        int tok = m0 + r + 8 * hi;       // < 208 always
        qp[(size_t)tok * HD] = (__bf16)(acc[r] + bias);
      }
    } else if (n0 < 128) {               // ---- K ----
      int d = c - 64;
      bf16_t* kp = Kb + (size_t)bh * NPAD * HD + d;
      #pragma unroll
      for (int r = 0; r < 8; ++r) {
        int tok = m0 + r + 8 * hi;
        kp[(size_t)tok * HD] = (__bf16)(acc[r] + bias);
      }
    } else {                             // ---- V (transposed, vectorized) ----
      int d = c - 128;
      v8bf pk;
      #pragma unroll
      for (int r = 0; r < 8; ++r) pk[r] = (__bf16)(acc[r] + bias);
      bf16_t* vp = VTb + ((size_t)bh * HD + d) * NPAD + m0 + 8 * hi;
      *(v8bf*)vp = pk;                   // 8 consecutive tokens, 16B aligned
    }
  }
}

// ---------------------------------------------------------------------------
// Kernel 2: flash attention per (b,h). One wave owns one 16-query tile.
//   Loop over 32 keys/step: 4 score WMMAs, online softmax (shfl_xor row
//   reductions within 16-lane halves), LDS C->A transpose of P, 4 PV WMMAs.
// ---------------------------------------------------------------------------
__global__ __launch_bounds__(128)
void attn_kernel(const bf16_t* __restrict__ Qb, const bf16_t* __restrict__ Kb,
                 const bf16_t* __restrict__ VTb, bf16_t* __restrict__ Ab) {
  __shared__ __bf16 plds[4][16][32];     // per-wave P-transpose staging (1KB/wave)

  int bh   = blockIdx.x;
  int b    = bh / HEADS;
  int h    = bh % HEADS;
  int lane = threadIdx.x & 31;
  int wave = threadIdx.x >> 5;
  int ln   = lane & 15;
  int hi   = lane >> 4;

  const bf16_t* qbase  = Qb  + (size_t)bh * NPAD * HD;
  const bf16_t* kbase  = Kb  + (size_t)bh * NPAD * HD;
  const bf16_t* vtbase = VTb + (size_t)bh * HD * NPAD;
  const float invScale = 0.03608439182435161f;   // 1/sqrt(768)

  for (int qt = wave; qt < QTILES; qt += 4) {
    int q0 = qt * 16;

    // Preload Q tile in A layout for both k-steps (bf16, contiguous 16B chunks)
    v16bf aq[2];
    #pragma unroll
    for (int s = 0; s < 2; ++s) {
      const bf16_t* qp = qbase + (size_t)(q0 + ln) * HD + s * 32 + 16 * hi;
      v8bf c0 = *(const v8bf*)qp;
      v8bf c1 = *(const v8bf*)(qp + 8);
      #pragma unroll
      for (int e = 0; e < 8; ++e) { aq[s][e] = c0[e]; aq[s][e + 8] = c1[e]; }
    }

    float mrow[8], lrow[8];
    v8f acc[4];
    #pragma unroll
    for (int r = 0; r < 8; ++r) { mrow[r] = -3.0e38f; lrow[r] = 0.0f; }
    #pragma unroll
    for (int t = 0; t < 4; ++t) { v8f z = {}; acc[t] = z; }

    for (int j0 = 0; j0 < NSEQ; j0 += 32) {
      // ---- scores: two 16-key subtiles, contraction over d=64 ----
      v8f sc[2];
      #pragma unroll
      for (int t = 0; t < 2; ++t) {
        v8f s_acc = {};
        int j  = j0 + 16 * t + ln;
        int jc = (j < NPAD) ? j : 0;     // clamp address; masked below
        #pragma unroll
        for (int s = 0; s < 2; ++s) {
          const bf16_t* kp = kbase + (size_t)jc * HD + s * 32 + 16 * hi;
          v16bf bk = *(const v16bf*)kp;  // 16 contiguous bf16 (32B)
          s_acc = wmma_bf16(aq[s], bk, s_acc);
        }
        bool valid = (j0 + 16 * t + ln) < NSEQ;
        #pragma unroll
        for (int r = 0; r < 8; ++r)
          sc[t][r] = valid ? (s_acc[r] * invScale) : -3.0e38f;
      }

      // ---- online softmax: row reductions across each 16-lane half ----
      #pragma unroll
      for (int r = 0; r < 8; ++r) {
        float v = fmaxf(sc[0][r], sc[1][r]);
        v = fmaxf(v, __shfl_xor(v, 1, 32));
        v = fmaxf(v, __shfl_xor(v, 2, 32));
        v = fmaxf(v, __shfl_xor(v, 4, 32));
        v = fmaxf(v, __shfl_xor(v, 8, 32));
        float newm = fmaxf(mrow[r], v);
        float corr = __expf(mrow[r] - newm);
        float p0v  = __expf(sc[0][r] - newm);
        float p1v  = __expf(sc[1][r] - newm);
        float srow = p0v + p1v;
        srow += __shfl_xor(srow, 1, 32);
        srow += __shfl_xor(srow, 2, 32);
        srow += __shfl_xor(srow, 4, 32);
        srow += __shfl_xor(srow, 8, 32);
        lrow[r] = lrow[r] * corr + srow;
        mrow[r] = newm;
        #pragma unroll
        for (int t = 0; t < 4; ++t) acc[t][r] *= corr;
        // stage P (C layout) into LDS for the A-layout transpose
        int row = r + 8 * hi;
        plds[wave][row][ln]      = (__bf16)p0v;
        plds[wave][row][ln + 16] = (__bf16)p1v;
      }
      __builtin_amdgcn_wave_barrier();   // same-wave DS ops are in-order

      // ---- read P back in A layout ----
      v16bf pa;
      {
        v8bf c0 = *(const v8bf*)&plds[wave][ln][16 * hi];
        v8bf c1 = *(const v8bf*)&plds[wave][ln][16 * hi + 8];
        #pragma unroll
        for (int e = 0; e < 8; ++e) { pa[e] = c0[e]; pa[e + 8] = c1[e]; }
      }
      __builtin_amdgcn_wave_barrier();

      // ---- P x V: 4 d-tiles, contraction over 32 keys ----
      int js = j0 + 16 * hi;
      int jv = (js < NPAD) ? js : 0;     // p is 0 beyond 197; V pad cols hold bias
      #pragma unroll
      for (int t = 0; t < 4; ++t) {
        const bf16_t* vp = vtbase + (size_t)(16 * t + ln) * NPAD + jv;
        v16bf bv = *(const v16bf*)vp;    // 16 contiguous bf16 (32B)
        acc[t] = wmma_bf16(pa, bv, acc[t]);
      }
    }

    // ---- finalize and store attn [b*197+tok][h*64 + d] bf16 ----
    #pragma unroll
    for (int r = 0; r < 8; ++r) {
      float inv = 1.0f / lrow[r];
      int tok = q0 + r + 8 * hi;
      if (tok < NSEQ) {
        size_t rowbase = ((size_t)b * NSEQ + tok) * FEATS + h * HD;
        #pragma unroll
        for (int t = 0; t < 4; ++t)
          Ab[rowbase + 16 * t + ln] = (__bf16)(acc[t][r] * inv);
      }
    }
  }
}

// ---------------------------------------------------------------------------
// Kernel 3: out = attn[25216,768] x W_out + b_out  (25216 = 788 * 32)
//   grid (788, 3) x 128 threads; wave w owns 64 cols x 32 rows; each B tile
//   load feeds 2 WMMAs (two row tiles).
// ---------------------------------------------------------------------------
__global__ __launch_bounds__(128)
void out_kernel(const bf16_t* __restrict__ Ab, const bf16_t* __restrict__ WT,
                const float* __restrict__ bias, float* __restrict__ out) {
  int rt   = blockIdx.x;                   // 32-row tile, 0..787
  int wave = threadIdx.x >> 5;
  int cg   = blockIdx.y * 4 + wave;        // col group 0..11 (64 cols each)
  int lane = threadIdx.x & 31;
  int ln   = lane & 15;
  int hi   = lane >> 4;
  int row0 = rt * 32;

  v8f acc[2][4];
  #pragma unroll
  for (int m = 0; m < 2; ++m)
    #pragma unroll
    for (int t = 0; t < 4; ++t) { v8f z = {}; acc[m][t] = z; }

  for (int kt = 0; kt < FEATS / 32; ++kt) {
    // A operands: two 16-row tiles (bf16, two 16B chunks per lane each)
    v16bf a[2];
    #pragma unroll
    for (int m = 0; m < 2; ++m) {
      const bf16_t* ap = Ab + (size_t)(row0 + 16 * m + ln) * FEATS
                            + kt * 32 + 16 * hi;
      v8bf c0 = *(const v8bf*)ap;
      v8bf c1 = *(const v8bf*)(ap + 8);
      #pragma unroll
      for (int e = 0; e < 8; ++e) { a[m][e] = c0[e]; a[m][e + 8] = c1[e]; }
    }

    #pragma unroll
    for (int t = 0; t < 4; ++t) {
      int col = cg * 64 + 16 * t + ln;
      const bf16_t* bp = WT + (size_t)col * FEATS + kt * 32 + 16 * hi;
      v16bf bv = *(const v16bf*)bp;        // contiguous 32B, L2-resident
      acc[0][t] = wmma_bf16(a[0], bv, acc[0][t]);
      acc[1][t] = wmma_bf16(a[1], bv, acc[1][t]);
    }
  }

  #pragma unroll
  for (int t = 0; t < 4; ++t) {
    int col = cg * 64 + 16 * t + ln;
    float bb = bias[col];
    #pragma unroll
    for (int m = 0; m < 2; ++m)
      #pragma unroll
      for (int r = 0; r < 8; ++r) {
        int row = row0 + 16 * m + r + 8 * hi;
        out[(size_t)row * FEATS + col] = acc[m][t][r] + bb;
      }
  }
}

// ---------------------------------------------------------------------------
extern "C" void kernel_launch(void* const* d_in, const int* in_sizes, int n_in,
                              void* d_out, int out_size, void* d_ws, size_t ws_size,
                              hipStream_t stream) {
  const float* x    = (const float*)d_in[0];   // [128,197,768]
  const float* Wq   = (const float*)d_in[1];   // [12,64,192]
  const float* bq   = (const float*)d_in[2];   // [12,192]
  const float* Wout = (const float*)d_in[3];   // [768,768]
  const float* bout = (const float*)d_in[4];   // [768]
  float* out = (float*)d_out;                  // [128,197,768] f32

  const size_t qkvElems  = (size_t)BATCH * HEADS * NPAD * HD;   // 20,447,232
  const size_t attnElems = (size_t)BATCH * NSEQ * FEATS;        // 19,365,888
  const size_t wtElems   = (size_t)FEATS * FEATS;               // 589,824

  char* ws = (char*)d_ws;
  bf16_t* Qb  = (bf16_t*)ws;
  bf16_t* Kb  = Qb  + qkvElems;
  bf16_t* VTb = Kb  + qkvElems;
  bf16_t* Ab  = VTb + qkvElems;
  bf16_t* WT  = Ab  + attnElems;
  bf16_t* WqT = WT  + wtElems;                 // [12,192,64] bf16

  wt_kernel <<<(FEATS * FEATS) / 256, 256, 0, stream>>>(Wout, WT);
  wqt_kernel<<<(HEADS * HD * 3 * HD) / 256, 256, 0, stream>>>(Wq, WqT);
  qkv_kernel<<<BATCH * HEADS, 128, 0, stream>>>(x, WqT, bq, Qb, Kb, VTb);
  attn_kernel<<<BATCH * HEADS, 128, 0, stream>>>(Qb, Kb, VTb, Ab);
  out_kernel<<<dim3((BATCH * NSEQ) / 32, 3), 128, 0, stream>>>(Ab, WT, bout, out);
}
